// RECT_L_13975823582298
// MI455X (gfx1250) — matrix operand order repaired and checked by reading, not compiled, not run on hardware
//
#include <hip/hip_runtime.h>
#include <hip/hip_bf16.h>

typedef __attribute__((ext_vector_type(4)))  float  f32x4;
typedef __attribute__((ext_vector_type(4)))  __bf16 v4bf;
typedef __attribute__((ext_vector_type(8)))  __bf16 v8bf;
typedef __attribute__((ext_vector_type(16))) __bf16 v16bf;
typedef __attribute__((ext_vector_type(8)))  float  v8f;

#define LDSS 40  // padded LDS row stride in bf16 elems (80B: 8B/16B aligned, spreads banks)

// ---------------------------------------------------------------------------
// Stage one 128x32 A tile (f32->bf16, row-major [m][k]) and one 32x128 B tile
// (f32->bf16, transposed into [n][k]) into LDS. A stores packed as b64.
// ---------------------------------------------------------------------------
template<bool NTA>
__device__ __forceinline__ void stage_tiles(
    const float* __restrict__ A, int lda,
    const float* __restrict__ B, int ldb,
    int M, int mBase, int nBase, int k0, int tid,
    const float* __restrict__ biasA,
    __bf16* __restrict__ sA, __bf16* __restrict__ sBt)
{
    #pragma unroll
    for (int i = 0; i < 4; ++i) {
        int idx = tid + i * 256;
        int r   = idx >> 3;
        int c4  = (idx & 7) << 2;
        int gr  = mBase + r;
        f32x4 v = {0.f, 0.f, 0.f, 0.f};
        if (gr < M) {
            const f32x4* p = (const f32x4*)(A + (size_t)gr * lda + k0 + c4);
            v = NTA ? __builtin_nontemporal_load(p) : *p;
            __builtin_prefetch((const float*)p + 32, 0, 1);  // next k-slab (speculative)
        }
        if (biasA) {
            v.x += biasA[k0 + c4 + 0];
            v.y += biasA[k0 + c4 + 1];
            v.z += biasA[k0 + c4 + 2];
            v.w += biasA[k0 + c4 + 3];
        }
        v4bf pk = { (__bf16)v.x, (__bf16)v.y, (__bf16)v.z, (__bf16)v.w };
        *(v4bf*)&sA[r * LDSS + c4] = pk;          // one ds_store_b64
    }
    #pragma unroll
    for (int i = 0; i < 4; ++i) {
        int idx = tid + i * 256;
        int kk  = idx >> 5;
        int n4  = (idx & 31) << 2;
        f32x4 v = *(const f32x4*)(B + (size_t)(k0 + kk) * ldb + nBase + n4);
        sBt[(n4 + 0) * LDSS + kk] = (__bf16)v.x;  // transposed -> scalar b16 stores
        sBt[(n4 + 1) * LDSS + kk] = (__bf16)v.y;
        sBt[(n4 + 2) * LDSS + kk] = (__bf16)v.z;
        sBt[(n4 + 3) * LDSS + kk] = (__bf16)v.w;
    }
}

// ---------------------------------------------------------------------------
// Tiled bf16-WMMA GEMM:  C[M,Ntot] = A[M,K](f32) * B[K,Ntot](f32)  (+biases)
// 256 threads = 8 waves, 128x128 tile, K-step 32, double-buffered LDS.
// NTA: stream A with non-temporal loads (don't pollute L2).
// NTC: stream C with non-temporal stores.
// Requires: K % 32 == 0, Ntot % 128 == 0.
// ---------------------------------------------------------------------------
template<bool NTA, bool NTC>
__global__ __launch_bounds__(256) void gcn_gemm_bf16_wmma(
    const float* __restrict__ A, int lda,
    const float* __restrict__ B, int ldb,
    float* __restrict__ C, int ldc,
    int M, int K,
    const float* __restrict__ biasA,
    const float* __restrict__ biasOut)
{
    __shared__ __bf16 sA [2][128 * LDSS];
    __shared__ __bf16 sBt[2][128 * LDSS];

    const int tid   = threadIdx.x;
    const int mBase = blockIdx.x * 128;
    const int nBase = blockIdx.y * 128;
    const int w     = tid >> 5;
    const int lane  = tid & 31;
    const int l16   = lane & 15;
    const int hi    = lane >> 4;

    v8f acc[8] = {};

    stage_tiles<NTA>(A, lda, B, ldb, M, mBase, nBase, 0, tid, biasA, sA[0], sBt[0]);
    __syncthreads();

    int buf = 0;
    for (int k0 = 0; k0 < K; k0 += 32) {
        if (k0 + 32 < K)  // prefetch next tiles into the other buffer
            stage_tiles<NTA>(A, lda, B, ldb, M, mBase, nBase, k0 + 32, tid,
                             biasA, sA[buf ^ 1], sBt[buf ^ 1]);

        // A fragment (16x32): m = l16; halves j -> k = hi*16 + j
        const __bf16* ap = &sA[buf][(w * 16 + l16) * LDSS + hi * 16];
        v8bf alo = *(const v8bf*)ap;
        v8bf ahi = *(const v8bf*)(ap + 8);
        v16bf a = __builtin_shufflevector(alo, ahi,
                    0,1,2,3,4,5,6,7,8,9,10,11,12,13,14,15);

        #pragma unroll
        for (int t = 0; t < 8; ++t) {
            // B fragment (32x16): n = l16; halves j -> k = hi*16 + j
            const __bf16* bp = &sBt[buf][(t * 16 + l16) * LDSS + hi * 16];
            v8bf blo = *(const v8bf*)bp;
            v8bf bhi = *(const v8bf*)(bp + 8);
            v16bf b = __builtin_shufflevector(blo, bhi,
                        0,1,2,3,4,5,6,7,8,9,10,11,12,13,14,15);
            acc[t] = __builtin_amdgcn_wmma_f32_16x16x32_bf16(
                         false, a, false, b, (short)0, acc[t], false, false);
        }
        __syncthreads();
        buf ^= 1;
    }

    // D layout: n = l16, VGPR r -> m = w*16 + hi*8 + r
    #pragma unroll
    for (int t = 0; t < 8; ++t) {
        int gn = nBase + t * 16 + l16;
        float bo = biasOut ? biasOut[gn] : 0.0f;
        #pragma unroll
        for (int r = 0; r < 8; ++r) {
            int gm = mBase + w * 16 + hi * 8 + r;
            if (gm < M) {
                float val = acc[t][r] + bo;
                float* cp = &C[(size_t)gm * ldc + gn];
                if (NTC) __builtin_nontemporal_store(val, cp);
                else     *cp = val;
            }
        }
    }
}

// ---------------------------------------------------------------------------
// Degree / normalization / aggregation kernels
// ---------------------------------------------------------------------------
__global__ __launch_bounds__(256) void gcn_deg_init(float* __restrict__ deg, int N) {
    int i = blockIdx.x * blockDim.x + threadIdx.x;
    if (i < N) deg[i] = 1.0f;  // self loop
}

__global__ __launch_bounds__(256) void gcn_deg_edges(const long long* __restrict__ col,
                                                     float* __restrict__ deg, int E) {
    int e = blockIdx.x * blockDim.x + threadIdx.x;
    if (e < E) {
        long long c = __builtin_nontemporal_load(&col[e]);  // streamed once
        __hip_atomic_fetch_add(&deg[(int)c], 1.0f,
                               __ATOMIC_RELAXED, __HIP_MEMORY_SCOPE_AGENT);
    }
}

__global__ __launch_bounds__(256) void gcn_deg_rsqrt(float* __restrict__ deg, int N) {
    int i = blockIdx.x * blockDim.x + threadIdx.x;
    if (i < N) deg[i] = rsqrtf(deg[i]);  // deg >= 1 always (self loop)
}

// agg init = self-loop contribution: agg[i][c] = h[i][c] * dis[i]^2
__global__ __launch_bounds__(256) void gcn_agg_init(const float* __restrict__ h,
                                                    const float* __restrict__ dis,
                                                    float* __restrict__ agg, int total) {
    int idx = blockIdx.x * blockDim.x + threadIdx.x;
    if (idx < total) {
        float d = dis[idx >> 7];  // HID = 128
        agg[idx] = h[idx] * d * d;
    }
}

// 128 threads per edge; h + agg (2 x 51.2 MB) are L2-resident (192 MB L2),
// so the random gather + atomic scatter stay on-chip.
__global__ __launch_bounds__(256) void gcn_agg_edges(const long long* __restrict__ row,
                                                     const long long* __restrict__ col,
                                                     const float* __restrict__ dis,
                                                     const float* __restrict__ h,
                                                     float* __restrict__ agg, int E) {
    int e = blockIdx.x * 2 + (threadIdx.x >> 7);
    int c = threadIdx.x & 127;
    if (e >= E) return;
    int s = (int)__builtin_nontemporal_load(&row[e]);
    int t = (int)__builtin_nontemporal_load(&col[e]);
    float nrm = dis[s] * dis[t];
    float val = h[(size_t)s * 128 + c] * nrm;
    __hip_atomic_fetch_add(&agg[(size_t)t * 128 + c], val,
                           __ATOMIC_RELAXED, __HIP_MEMORY_SCOPE_AGENT);
}

// ---------------------------------------------------------------------------
extern "C" void kernel_launch(void* const* d_in, const int* in_sizes, int n_in,
                              void* d_out, int out_size, void* d_ws, size_t ws_size,
                              hipStream_t stream) {
    const float*     x  = (const float*)d_in[0];
    const long long* ei = (const long long*)d_in[1];  // [2, E] int64
    const float*     Wc = (const float*)d_in[2];      // [INC, HID]
    const float*     bc = (const float*)d_in[3];      // [HID]
    const float*     Wl = (const float*)d_in[4];      // [HID, INC]
    const float*     bl = (const float*)d_in[5];      // [INC]

    const int HID = in_sizes[3];            // 128
    const int INC = in_sizes[2] / HID;      // 512
    const int N   = in_sizes[0] / INC;      // 100000
    const int E   = in_sizes[1] / 2;        // 1600000

    // Workspace: deg/dis (N f32) + agg (N*HID f32). h staged in front of d_out
    // (dead before GEMM2 overwrites d_out).
    float* deg = (float*)d_ws;
    size_t degBytes = (((size_t)N * 4) + 255) & ~(size_t)255;
    float* agg = (float*)((char*)d_ws + degBytes);
    float* h   = (float*)d_out;
    float* out = (float*)d_out;

    const long long* row = ei;
    const long long* col = ei + E;

    dim3 blk(256);

    gcn_deg_init <<<(N + 255) / 256, blk, 0, stream>>>(deg, N);
    gcn_deg_edges<<<(E + 255) / 256, blk, 0, stream>>>(col, deg, E);
    gcn_deg_rsqrt<<<(N + 255) / 256, blk, 0, stream>>>(deg, N);

    // GEMM1: h[N,HID] = x[N,INC] @ Wc[INC,HID]; x streamed NT, h kept in L2
    dim3 g1((N + 127) / 128, HID / 128);
    gcn_gemm_bf16_wmma<true, false><<<g1, blk, 0, stream>>>(
        x, INC, Wc, HID, h, HID, N, INC, nullptr, nullptr);

    int total = N * HID;
    gcn_agg_init <<<(total + 255) / 256, blk, 0, stream>>>(h, deg, agg, total);
    gcn_agg_edges<<<(E + 1) / 2,        blk, 0, stream>>>(row, col, deg, h, agg, E);

    // GEMM2: out[N,INC] = (agg + bc)[N,HID] @ Wl[HID,INC] + bl; out streamed NT
    dim3 g2((N + 127) / 128, INC / 128);
    gcn_gemm_bf16_wmma<false, true><<<g2, blk, 0, stream>>>(
        agg, HID, Wl, INC, out, INC, N, HID, bc, bl);
}
